// ChunkGatedDeltaModel_33328946217460
// MI455X (gfx1250) — matrix operand order repaired and checked by reading, not compiled, not run on hardware
//
#include <hip/hip_runtime.h>

namespace {

constexpr int B_  = 4;
constexpr int H_  = 16;
constexpr int T_  = 2048;
constexpr int DK  = 128;   // key dim
constexpr int DV  = 128;   // value dim
constexpr int CS  = 64;    // time-chunk length
constexpr int VC  = 64;    // V columns per block (2-way V split)
constexpr int NCH = T_ / CS;

typedef __bf16 v16bf __attribute__((ext_vector_type(16)));
typedef float  v8f   __attribute__((ext_vector_type(8)));

union FragBF { v16bf v; unsigned short u[16]; };
union FragF  { v8f   v; float f[8]; };

__device__ __forceinline__ unsigned short f2bfu(float x){
  union { float f; unsigned int u; } c; c.f = x;
  unsigned int r = (c.u + 0x7FFFu + ((c.u >> 16) & 1u)) >> 16;  // RNE
  return (unsigned short)r;
}
__device__ __forceinline__ float bf2f(unsigned short s){
  union { unsigned int u; float f; } c; c.u = ((unsigned int)s) << 16;
  return c.f;
}

// --- ISA 7.12.2 wave32 fragment index maps -------------------------------
// A (16x32, 16-bit): lane m=lane&15; lanes<16 hold K 0-7,16-23; lanes>=16 hold 8-15,24-31
__device__ __forceinline__ int akIdx(int e, int lane){
  int kb = (lane & 16) ? 8 : 0;
  return (e < 8) ? (kb + e) : (16 + kb + (e - 8));
}
// B (32x16, 16-bit): lane n=lane&15; lanes<16 hold K 0-7,16-23; lanes>=16 hold 8-15,24-31
__device__ __forceinline__ int bkIdx(int e, int lane){
  int kb = (lane & 16) ? 8 : 0;
  return (e < 8) ? (kb + e) : (16 + kb + (e - 8));
}

__device__ __forceinline__ void mma(FragF& c, const FragBF& a, const FragBF& b){
  c.v = __builtin_amdgcn_wmma_f32_16x16x32_bf16(false, a.v, false, b.v,
                                                (short)0, c.v, false, false);
}

__device__ __forceinline__ FragF zeroF(){
  FragF c;
  #pragma unroll
  for (int r = 0; r < 8; ++r) c.f[r] = 0.f;
  return c;
}

// ---- loaders ------------------------------------------------------------
__device__ __forceinline__ FragBF ldA_bf(const unsigned short* p, int ld,
                                         int m0, int k0, int lane){
  FragBF f; int m = m0 + (lane & 15);
  #pragma unroll
  for (int e = 0; e < 16; ++e) f.u[e] = p[m * ld + k0 + akIdx(e, lane)];
  return f;
}
__device__ __forceinline__ FragBF ldA_bf_rs(const unsigned short* p, int ld,
                                            int m0, int k0, int lane,
                                            const float* s){   // row scale s[m]
  FragBF f; int m = m0 + (lane & 15); float sc = s[m];
  #pragma unroll
  for (int e = 0; e < 16; ++e)
    f.u[e] = f2bfu(bf2f(p[m * ld + k0 + akIdx(e, lane)]) * sc);
  return f;
}
__device__ __forceinline__ FragBF ldA_f32(const float* p, int ld,
                                          int m0, int k0, int lane){
  FragBF f; int m = m0 + (lane & 15);
  #pragma unroll
  for (int e = 0; e < 16; ++e) f.u[e] = f2bfu(p[m * ld + k0 + akIdx(e, lane)]);
  return f;
}
// A[m][kk] = p[kk][m] * s[kk]   (transposed, k-scaled: K_dec^T)
__device__ __forceinline__ FragBF ldA_kT(const unsigned short* p, int ld,
                                         int m0, int k0, int lane,
                                         const float* s){
  FragBF f; int m = m0 + (lane & 15);
  #pragma unroll
  for (int e = 0; e < 16; ++e){
    int kk = k0 + akIdx(e, lane);
    f.u[e] = f2bfu(bf2f(p[kk * ld + m]) * s[kk]);
  }
  return f;
}
__device__ __forceinline__ FragBF ldB_bf(const unsigned short* p, int ld,
                                         int k0, int n0, int lane){
  FragBF f; int n = n0 + (lane & 15);
  #pragma unroll
  for (int e = 0; e < 16; ++e) f.u[e] = p[(k0 + bkIdx(e, lane)) * ld + n];
  return f;
}
// B = P^T, P row-major: B[k][n] = P[n][k]
__device__ __forceinline__ FragBF ldB_bfT(const unsigned short* p, int ld,
                                          int k0, int n0, int lane){
  FragBF f; int n = n0 + (lane & 15);
  #pragma unroll
  for (int e = 0; e < 16; ++e) f.u[e] = p[n * ld + k0 + bkIdx(e, lane)];
  return f;
}
__device__ __forceinline__ FragBF ldB_bf_ks(const unsigned short* p, int ld,
                                            int k0, int n0, int lane,
                                            const float* s){  // k-row scale
  FragBF f; int n = n0 + (lane & 15);
  #pragma unroll
  for (int e = 0; e < 16; ++e){
    int kk = k0 + bkIdx(e, lane);
    f.u[e] = f2bfu(bf2f(p[kk * ld + n]) * s[kk]);
  }
  return f;
}
__device__ __forceinline__ FragBF ldB_f32(const float* p, int ld,
                                          int k0, int n0, int lane){
  FragBF f; int n = n0 + (lane & 15);
  #pragma unroll
  for (int e = 0; e < 16; ++e) f.u[e] = f2bfu(p[(k0 + bkIdx(e, lane)) * ld + n]);
  return f;
}
__device__ __forceinline__ FragBF ldB_f32_plusI(const float* p, int ld,
                                                int k0, int n0, int lane){
  FragBF f; int n = n0 + (lane & 15);
  #pragma unroll
  for (int e = 0; e < 16; ++e){
    int kk = k0 + bkIdx(e, lane);
    float x = p[kk * ld + n] + ((kk == n) ? 1.0f : 0.0f);
    f.u[e] = f2bfu(x);
  }
  return f;
}
// C/D (f32 16x16): lane n=lane&15; VGPR r -> M = r + (lane<16 ? 0 : 8)
__device__ __forceinline__ FragF ldC_f32(const float* p, int ld, int m0, int n0,
                                         int lane, float scale){
  FragF c; int n = n0 + (lane & 15); int mb = m0 + ((lane & 16) ? 8 : 0);
  #pragma unroll
  for (int r = 0; r < 8; ++r) c.f[r] = p[(mb + r) * ld + n] * scale;
  return c;
}
__device__ __forceinline__ void stC_f32(float* p, int ld, int m0, int n0,
                                        int lane, const FragF& c){
  int n = n0 + (lane & 15); int mb = m0 + ((lane & 16) ? 8 : 0);
  #pragma unroll
  for (int r = 0; r < 8; ++r) p[(mb + r) * ld + n] = c.f[r];
}

__global__ __launch_bounds__(256, 1) void gated_delta_chunk_kernel(
    const float* __restrict__ qg, const float* __restrict__ kg,
    const float* __restrict__ vg, const float* __restrict__ gg_,
    const float* __restrict__ bg, const float* __restrict__ s0g,
    float* __restrict__ og, float* __restrict__ sfg)
{
  // ---- LDS (~201 KB of the 320 KB WGP pool) ----
  __shared__ float S[DK * VC];                 // running state, f32
  __shared__ unsigned short qs [CS * DK];      // normalized q * K^-1/2 (bf16)
  __shared__ unsigned short ks [CS * DK];      // normalized k (bf16)
  __shared__ unsigned short kbs[CS * DK];      // normalized k * beta (bf16)
  __shared__ unsigned short vbs[CS * VC];      // v * beta (bf16)
  __shared__ float Wn[CS * DK];                // -(T @ (beta e^b (x) k))
  __shared__ float Ub[CS * VC];                // U = T@(beta v) + Wn@S0
  __shared__ float Mb[4][CS * CS];             // L, Q, Tcur, Tnext
  __shared__ float gsh[CS], bsh[CS], bta[CS];
  __shared__ float eb[CS], ebi[CS], elo[CS];

  const int tid  = threadIdx.x;
  const int lane = tid & 31;
  const int w    = tid >> 5;
  const int bh   = blockIdx.x >> 1;
  const int vh   = (blockIdx.x & 1) * VC;

  // initial state -> LDS
  for (int i = tid; i < DK * VC; i += 256)
    S[i] = s0g[(size_t)bh * DK * DV + (size_t)(i / VC) * DV + vh + (i % VC)];
  __syncthreads();

  const float qscale = 0.08838834764831845f;   // 128^-0.5

  for (int c = 0; c < NCH; ++c){
    const int t0 = c * CS;

    // ---- stage per-step scalars ----
    if (tid < CS){
      gsh[tid] = gg_[(size_t)bh * T_ + t0 + tid];
      bta[tid] = bg [(size_t)bh * T_ + t0 + tid];
    }
    __syncthreads();

    if (c + 1 < NCH){   // prefetch next chunk (global_prefetch_b8)
      size_t nb = ((size_t)bh * T_ + t0 + CS) * DK;
      __builtin_prefetch(qg + nb + (size_t)tid * 32, 0, 1);
      __builtin_prefetch(kg + nb + (size_t)tid * 32, 0, 1);
      __builtin_prefetch(vg + ((size_t)bh * T_ + t0 + CS) * DV + (size_t)tid * 32, 0, 1);
    }

    // inclusive cumsum of gates (tiny, one lane)
    if (tid == 0){
      float a = 0.f;
      for (int i = 0; i < CS; ++i){ a += gsh[i]; bsh[i] = a; }
    }

    // ---- load + L2-normalize chunk rows, stage bf16 operands ----
    #pragma unroll
    for (int rr = 0; rr < 8; ++rr){
      int row = w * 8 + rr;
      size_t base = ((size_t)bh * T_ + t0 + row) * DK;
      float qv[4], kv[4]; float sq = 0.f, sk = 0.f;
      #pragma unroll
      for (int e = 0; e < 4; ++e){
        int col = lane + e * 32;
        qv[e] = qg[base + col]; kv[e] = kg[base + col];
        sq += qv[e] * qv[e];    sk += kv[e] * kv[e];
      }
      #pragma unroll
      for (int off = 16; off; off >>= 1){
        sq += __shfl_xor(sq, off, 32);
        sk += __shfl_xor(sk, off, 32);
      }
      float qiv = qscale / fmaxf(sqrtf(sq), 1e-12f);
      float kiv = 1.0f   / fmaxf(sqrtf(sk), 1e-12f);
      float bt  = bta[row];
      #pragma unroll
      for (int e = 0; e < 4; ++e){
        int col = lane + e * 32;
        qs [row * DK + col] = f2bfu(qv[e] * qiv);
        ks [row * DK + col] = f2bfu(kv[e] * kiv);
        kbs[row * DK + col] = f2bfu(kv[e] * kiv * bt);
      }
      size_t vbase = ((size_t)bh * T_ + t0 + row) * DV + vh;
      #pragma unroll
      for (int e = 0; e < 2; ++e){
        int col = lane + e * 32;
        vbs[row * VC + col] = f2bfu(vg[vbase + col] * bt);
      }
    }
    __syncthreads();
    if (tid < CS){
      float bi = bsh[tid];
      eb [tid] = __expf(bi);
      ebi[tid] = __expf(-bi);
      elo[tid] = __expf(bsh[CS - 1] - bi);
    }
    __syncthreads();

    float* Lc = Mb[0]; float* Qb = Mb[1]; float* Tc = Mb[2]; float* Tn = Mb[3];

    // ---- phase 1: L = strict_tril( (beta k) k^T * e^{b_i-b_j} ) ----
    for (int t = w; t < 16; t += 8){
      int mt = (t >> 2) * 16, nt = (t & 3) * 16;
      FragF cf = zeroF();
      #pragma unroll
      for (int k0 = 0; k0 < DK; k0 += 32){
        FragBF a = ldA_bf (kbs, DK, mt, k0, lane);
        FragBF b = ldB_bfT(ks , DK, k0, nt, lane);
        mma(cf, a, b);
      }
      int n = nt + (lane & 15); int mb = mt + ((lane & 16) ? 8 : 0);
      float ebn = ebi[n];                       // hoisted, unconditional
      #pragma unroll
      for (int r = 0; r < 8; ++r){
        int i = mb + r;
        float val = cf.f[r] * eb[i] * ebn;      // unconditional loads -> cndmask
        Lc[i * CS + n] = (n < i) ? val : 0.f;
      }
    }
    __syncthreads();

    // Tc = I - L
    for (int i = tid; i < CS * CS; i += 256){
      int r = i / CS, cc = i % CS;
      Tc[i] = ((r == cc) ? 1.f : 0.f) - Lc[i];
    }
    __syncthreads();

    // ---- phase 2: (I+L)^{-1} = (I-L)(I+L^2)(I+L^4)(I+L^8)(I+L^16)(I+L^32)
    for (int s = 0; s < 5; ++s){
      for (int t = w; t < 16; t += 8){          // Qb = Lc @ Lc
        int mt = (t >> 2) * 16, nt = (t & 3) * 16;
        FragF cf = zeroF();
        #pragma unroll
        for (int k0 = 0; k0 < CS; k0 += 32){
          FragBF a = ldA_f32(Lc, CS, mt, k0, lane);
          FragBF b = ldB_f32(Lc, CS, k0, nt, lane);
          mma(cf, a, b);
        }
        stC_f32(Qb, CS, mt, nt, lane, cf);
      }
      __syncthreads();
      for (int t = w; t < 16; t += 8){          // Tn = Tc @ (I + Qb)
        int mt = (t >> 2) * 16, nt = (t & 3) * 16;
        FragF cf = zeroF();
        #pragma unroll
        for (int k0 = 0; k0 < CS; k0 += 32){
          FragBF a = ldA_f32(Tc, CS, mt, k0, lane);
          FragBF b = ldB_f32_plusI(Qb, CS, k0, nt, lane);
          mma(cf, a, b);
        }
        stC_f32(Tn, CS, mt, nt, lane, cf);
      }
      __syncthreads();
      float* tp = Lc; Lc = Qb; Qb = tp;
      tp = Tc; Tc = Tn; Tn = tp;
    }

    // ---- phase 3: Wn = -(Tc @ (e^b beta k)),  Ub = Tc @ (beta v) ----
    for (int t = w; t < 32; t += 8){
      int mt = (t >> 3) * 16, nt = (t & 7) * 16;
      FragF cf = zeroF();
      #pragma unroll
      for (int k0 = 0; k0 < CS; k0 += 32){
        FragBF a = ldA_f32 (Tc, CS, mt, k0, lane);
        FragBF b = ldB_bf_ks(kbs, DK, k0, nt, lane, eb);
        mma(cf, a, b);
      }
      int n = nt + (lane & 15); int mb = mt + ((lane & 16) ? 8 : 0);
      #pragma unroll
      for (int r = 0; r < 8; ++r) Wn[(mb + r) * DK + n] = -cf.f[r];
    }
    for (int t = w; t < 16; t += 8){
      int mt = (t >> 2) * 16, nt = (t & 3) * 16;
      FragF cf = zeroF();
      #pragma unroll
      for (int k0 = 0; k0 < CS; k0 += 32){
        FragBF a = ldA_f32(Tc, CS, mt, k0, lane);
        FragBF b = ldB_bf (vbs, VC, k0, nt, lane);
        mma(cf, a, b);
      }
      stC_f32(Ub, VC, mt, nt, lane, cf);
    }
    __syncthreads();

    // ---- phase 4: Ub += Wn @ S0 ----
    for (int t = w; t < 16; t += 8){
      int mt = (t >> 2) * 16, nt = (t & 3) * 16;
      FragF cf = ldC_f32(Ub, VC, mt, nt, lane, 1.0f);
      #pragma unroll
      for (int k0 = 0; k0 < DK; k0 += 32){
        FragBF a = ldA_f32(Wn, DK, mt, k0, lane);
        FragBF b = ldB_f32(S , VC, k0, nt, lane);
        mma(cf, a, b);
      }
      stC_f32(Ub, VC, mt, nt, lane, cf);
    }
    __syncthreads();

    // ---- phase 5: Oreg = (e^b q') @ S0  (kept in registers) ----
    FragF oacc[2];
    #pragma unroll
    for (int it = 0; it < 2; ++it){
      int t = w + it * 8;
      int mt = (t >> 2) * 16, nt = (t & 3) * 16;
      FragF cf = zeroF();
      #pragma unroll
      for (int k0 = 0; k0 < DK; k0 += 32){
        FragBF a = ldA_bf_rs(qs, DK, mt, k0, lane, eb);
        FragBF b = ldB_f32  (S , VC, k0, nt, lane);
        mma(cf, a, b);
      }
      oacc[it] = cf;
    }

    // ---- phase 6: AT = tril_incl( q' k^T * e^{b_i-b_j} ) into free buffer
    float* ATb = Tn;
    for (int t = w; t < 16; t += 8){
      int mt = (t >> 2) * 16, nt = (t & 3) * 16;
      FragF cf = zeroF();
      #pragma unroll
      for (int k0 = 0; k0 < DK; k0 += 32){
        FragBF a = ldA_bf (qs, DK, mt, k0, lane);
        FragBF b = ldB_bfT(ks, DK, k0, nt, lane);
        mma(cf, a, b);
      }
      int n = nt + (lane & 15); int mb = mt + ((lane & 16) ? 8 : 0);
      float ebn = ebi[n];                       // hoisted, unconditional
      #pragma unroll
      for (int r = 0; r < 8; ++r){
        int i = mb + r;
        float val = cf.f[r] * eb[i] * ebn;
        ATb[i * CS + n] = (n <= i) ? val : 0.f;
      }
    }
    __syncthreads();

    // ---- phase 7: O = Oreg + AT @ U  -> global ----
    #pragma unroll
    for (int it = 0; it < 2; ++it){
      int t = w + it * 8;
      int mt = (t >> 2) * 16, nt = (t & 3) * 16;
      FragF cf = oacc[it];
      #pragma unroll
      for (int k0 = 0; k0 < CS; k0 += 32){
        FragBF a = ldA_f32(ATb, CS, mt, k0, lane);
        FragBF b = ldB_f32(Ub , VC, k0, nt, lane);
        mma(cf, a, b);
      }
      int n = vh + nt + (lane & 15); int mb = mt + ((lane & 16) ? 8 : 0);
      #pragma unroll
      for (int r = 0; r < 8; ++r)
        og[((size_t)bh * T_ + t0 + mb + r) * DV + n] = cf.f[r];
    }

    // ---- phase 8: S = e^{b_last} S + (e^{b_last-b} k)^T @ U ----
    {
      float eblast = eb[CS - 1];
      for (int t = w; t < 32; t += 8){
        int mt = (t >> 2) * 16, nt = (t & 3) * 16;
        FragF cf = ldC_f32(S, VC, mt, nt, lane, eblast);
        #pragma unroll
        for (int k0 = 0; k0 < CS; k0 += 32){
          FragBF a = ldA_kT (ks, DK, mt, k0, lane, elo);
          FragBF b = ldB_f32(Ub, VC, k0, nt, lane);
          mma(cf, a, b);
        }
        stC_f32(S, VC, mt, nt, lane, cf);
      }
    }
    __syncthreads();
  }

  // ---- final state -> global ----
  for (int i = tid; i < DK * VC; i += 256)
    sfg[(size_t)bh * DK * DV + (size_t)(i / VC) * DV + vh + (i % VC)] = S[i];
}

} // namespace

extern "C" void kernel_launch(void* const* d_in, const int* in_sizes, int n_in,
                              void* d_out, int out_size, void* d_ws, size_t ws_size,
                              hipStream_t stream){
  const float* q  = (const float*)d_in[0];
  const float* k  = (const float*)d_in[1];
  const float* v  = (const float*)d_in[2];
  const float* g  = (const float*)d_in[3];
  const float* b  = (const float*)d_in[4];
  const float* s0 = (const float*)d_in[5];
  float* o  = (float*)d_out;
  float* sf = o + (size_t)B_ * H_ * T_ * DV;

  dim3 grid(B_ * H_ * (DV / VC));   // 128 blocks: (b,h) x V-half
  gated_delta_chunk_kernel<<<grid, 256, 0, stream>>>(q, k, v, g, b, s0, o, sf);

  (void)in_sizes; (void)n_in; (void)d_out; (void)out_size;
  (void)d_ws; (void)ws_size;
}
